// RGCN_layer_28707561406962
// MI455X (gfx1250) — matrix operand order repaired and checked by reading, not compiled, not run on hardware
//
#include <hip/hip_runtime.h>
#include <hip/hip_bf16.h>

typedef __attribute__((ext_vector_type(16))) __bf16 v16bf;
typedef __attribute__((ext_vector_type(8)))  __bf16 v8bf;
typedef __attribute__((ext_vector_type(4)))  __bf16 v4bf;
typedef __attribute__((ext_vector_type(8)))  float  v8f;

#define D 128             // feature dim (D_IN == D_OUT == 128)
#define NT 4              // edge types
#define LDS_STRIDE (D + 8)   // W tile: 272B rows, 16B aligned, bank-friendly
#define A_STRIDE  (D + 8)    // A tile rows padded the same way
#define TM 32                // rows per GEMM iteration (2 wmma m-tiles)

// ---------------------------------------------------------------------------
__global__ void rgcn_zero_i32(int* __restrict__ p, int n) {
    int i = blockIdx.x * blockDim.x + threadIdx.x;
    if (i < n) p[i] = 0;
}

// ---------------------------------------------------------------------------
// one-shot f32 -> bf16 conversion of x (so GEMM A-stream is pure bf16)
__global__ void rgcn_cvt_bf16(const float* __restrict__ x,
                              __bf16* __restrict__ xb, int n4) {
    int i = blockIdx.x * blockDim.x + threadIdx.x;
    if (i < n4) {
        const float4 v = ((const float4*)x)[i];
        v4bf o;
        o[0] = (__bf16)v.x; o[1] = (__bf16)v.y;
        o[2] = (__bf16)v.z; o[3] = (__bf16)v.w;
        ((v4bf*)xb)[i] = o;
    }
}

// ---------------------------------------------------------------------------
// per-(etype,node) in-degree counts
__global__ void rgcn_count(const int* __restrict__ dst, int* __restrict__ cnt,
                           int total, int n_nodes, int n_edges) {
    int i = blockIdx.x * blockDim.x + threadIdx.x;
    if (i < total) {
        int t = i / n_edges;
        int d = dst[i];
        atomicAdd(&cnt[(size_t)t * n_nodes + d], 1);
    }
}

// ---------------------------------------------------------------------------
// async global->LDS copy of one 32x128 bf16 A tile (contiguous 8KB in global).
// 256 threads x 2 x b128. Tracked by ASYNCcnt.
__device__ __forceinline__ void issue_a_tile(const __bf16* __restrict__ gsrc,
                                             __bf16 (* __restrict__ At)[A_STRIDE],
                                             int tid) {
    const int r  = tid >> 3;           // 0..31 row
    const int cb = (tid & 7) * 16;     // element offset within row
    const __bf16* g0 = gsrc + r * D + cb;
    unsigned l0 = (unsigned)(size_t)&At[r][cb];
    unsigned l1 = (unsigned)(size_t)&At[r][cb + 8];
    asm volatile("global_load_async_to_lds_b128 %0, %1, off"
                 :: "v"(l0), "v"((unsigned long long)(size_t)g0) : "memory");
    asm volatile("global_load_async_to_lds_b128 %0, %1, off"
                 :: "v"(l1), "v"((unsigned long long)(size_t)(g0 + 8)) : "memory");
}

__device__ __forceinline__ void wait_async0() {
    asm volatile("s_wait_asynccnt 0x0" ::: "memory");
}

// ---------------------------------------------------------------------------
// GEMM: grid.y = t in [0,4]. t<4: wx[t] = bf16(x @ W[t] + b[t]) -> workspace
//                           t==4: out   = x @ W_self + b_self (f32)
// 256 threads = 8 waves; wave w owns N-columns [16w,16w+16).
// W_t transposed in LDS (bf16), B-frags built once per block.
// A tiles (32 rows) double-buffered in LDS via async global->LDS loads,
// prefetched one iteration ahead. Per iteration: 16x ds_load_b128 up front,
// then 8x v_wmma_f32_16x16x32_bf16 as two interleaved accumulator chains.
__global__ __launch_bounds__(256) void rgcn_gemm(
    const __bf16* __restrict__ xb,      // [N,128] bf16 (workspace)
    const float* __restrict__ W,        // [4,128,128]
    const float* __restrict__ bvec,     // [4,128]
    const float* __restrict__ W_self,   // [128,128]
    const float* __restrict__ b_self,   // [128]
    __bf16* __restrict__ wx,            // [4,N,128] bf16 (workspace)
    float* __restrict__ out,            // [N,128]
    int n_nodes, int n_groups)          // n_groups = N / 32
{
    const int t = blockIdx.y;
    const float* Wt = (t < NT) ? (W + (size_t)t * D * D) : W_self;
    const float* bt = (t < NT) ? (bvec + (size_t)t * D) : b_self;

    __shared__ __bf16 Wl[D][LDS_STRIDE];        // transposed W: Wl[n][k]
    __shared__ __bf16 Atile[2][TM][A_STRIDE];   // double-buffered A tiles
    __shared__ float  bias_s[D];

    // prefetch first A tile for this block (overlaps with W staging)
    issue_a_tile(xb + (size_t)blockIdx.x * TM * D, Atile[0], threadIdx.x);

    // stage W (transposed, bf16) + bias in LDS
    for (int i = threadIdx.x; i < D * D; i += blockDim.x) {
        int k = i >> 7, n = i & (D - 1);
        Wl[n][k] = (__bf16)Wt[k * D + n];
    }
    if (threadIdx.x < D) bias_s[threadIdx.x] = bt[threadIdx.x];
    __syncthreads();

    const int lane = threadIdx.x & 31;
    const int half = lane >> 4;          // 0: lanes 0-15, 1: lanes 16-31
    const int lm   = lane & 15;
    const int n0   = (threadIdx.x >> 5) * 16;

    // B fragments (32x16 bf16, KxN): lane half h holds K = 16h..16h+15
    v16bf bfrag[4];
    #pragma unroll
    for (int kk = 0; kk < 4; ++kk) {
        int kb = kk * 32 + half * 16;
        #pragma unroll
        for (int e = 0; e < 16; ++e)
            bfrag[kk][e] = Wl[n0 + lm][kb + e];
    }
    const float mybias = bias_s[n0 + lm];

    int parity = 0;
    for (int grp = blockIdx.x; grp < n_groups; grp += gridDim.x) {
        wait_async0();        // my async b128s for Atile[parity] landed
        __syncthreads();      // everyone's landed; prev reads of other buf done

        const int nxt = grp + gridDim.x;
        if (nxt < n_groups)   // prefetch next tile into the other buffer
            issue_a_tile(xb + (size_t)nxt * TM * D, Atile[parity ^ 1],
                         threadIdx.x);

        // load ALL A fragments for both m-tiles first (16x ds_load_b128),
        // A layout (16x32 bf16, MxK): lane = row lm; half h holds
        // K = 16h..16h+15 within each 32-wide K step
        v16bf afrag[2][4];
        #pragma unroll
        for (int mt = 0; mt < 2; ++mt) {
            #pragma unroll
            for (int kk = 0; kk < 4; ++kk) {
                const __bf16* ar = &Atile[parity][mt * 16 + lm]
                                         [kk * 32 + half * 16];
                const v8bf lo = *(const v8bf*)ar;
                const v8bf hi = *(const v8bf*)(ar + 8);
                #pragma unroll
                for (int e = 0; e < 8; ++e) {
                    afrag[mt][kk][e]     = lo[e];
                    afrag[mt][kk][8 + e] = hi[e];
                }
            }
        }

        // two independent accumulator chains, interleaved to hide the
        // WMMA->WMMA RAW latency on the XDL pipe
        v8f c0 = {}, c1 = {};
        #pragma unroll
        for (int kk = 0; kk < 4; ++kk) {
            c0 = __builtin_amdgcn_wmma_f32_16x16x32_bf16(
                     false, afrag[0][kk], false, bfrag[kk],
                     (short)0, c0, false, false);
            c1 = __builtin_amdgcn_wmma_f32_16x16x32_bf16(
                     false, afrag[1][kk], false, bfrag[kk],
                     (short)0, c1, false, false);
        }

        // C/D layout: VGPR r -> M = r + 8*half, N = lm
        const int row0 = grp * TM;
        if (t < NT) {
            __bf16* dp0 = wx + ((size_t)t * n_nodes + row0) * D + n0 + lm;
            __bf16* dp1 = dp0 + (size_t)16 * D;
            #pragma unroll
            for (int r = 0; r < 8; ++r) {
                dp0[(size_t)(r + half * 8) * D] = (__bf16)(c0[r] + mybias);
                dp1[(size_t)(r + half * 8) * D] = (__bf16)(c1[r] + mybias);
            }
        } else {
            float* dp0 = out + (size_t)row0 * D + n0 + lm;
            float* dp1 = dp0 + (size_t)16 * D;
            #pragma unroll
            for (int r = 0; r < 8; ++r) {
                dp0[(size_t)(r + half * 8) * D] = c0[r] + mybias;
                dp1[(size_t)(r + half * 8) * D] = c1[r] + mybias;
            }
        }
        parity ^= 1;
    }
}

// ---------------------------------------------------------------------------
// Scatter: one wave32 per (etype,edge); lane handles 4 features.
// out[dst] += wx[t][src] / cnt[t][dst]  (mean folded into per-edge scale)
__global__ __launch_bounds__(256) void rgcn_scatter(
    const __bf16* __restrict__ wx, const int* __restrict__ src,
    const int* __restrict__ dst, const int* __restrict__ cnt,
    float* __restrict__ out, int total, int n_nodes, int n_edges)
{
    int wid  = (blockIdx.x * blockDim.x + threadIdx.x) >> 5;  // wave-uniform
    int lane = threadIdx.x & 31;
    if (wid >= total) return;
    int t = wid / n_edges;
    int e = wid - t * n_edges;
    size_t eoff = (size_t)t * n_edges + e;
    int s = src[eoff];
    int d = dst[eoff];
    float inv = 1.0f / (float)cnt[(size_t)t * n_nodes + d];  // >=1: edge exists

    const v4bf mv = *(const v4bf*)(wx + ((size_t)t * n_nodes + s) * D + lane * 4);
    float* o = out + (size_t)d * D + lane * 4;
    #pragma unroll
    for (int i = 0; i < 4; ++i)
        atomicAdd(o + i, (float)mv[i] * inv);
}

// ---------------------------------------------------------------------------
extern "C" void kernel_launch(void* const* d_in, const int* in_sizes, int n_in,
                              void* d_out, int out_size, void* d_ws, size_t ws_size,
                              hipStream_t stream) {
    const float* x      = (const float*)d_in[0];   // [N,128]
    const float* W      = (const float*)d_in[1];   // [4,128,128]
    const float* b      = (const float*)d_in[2];   // [4,128]
    const float* W_self = (const float*)d_in[3];   // [128,128]
    const float* b_self = (const float*)d_in[4];   // [128]
    const int*   src    = (const int*)d_in[5];     // [4,E]
    const int*   dst    = (const int*)d_in[6];     // [4,E]
    float* out = (float*)d_out;

    const int n_nodes = in_sizes[0] / D;           // 200000
    const int n_edges = in_sizes[5] / NT;          // 500000

    // workspace: [cnt: 4*N int][wx: 4*N*128 bf16][xb: N*128 bf16]
    char* ws = (char*)d_ws;
    int*    cnt = (int*)ws;
    __bf16* wx  = (__bf16*)(ws + (size_t)NT * n_nodes * sizeof(int));
    __bf16* xb  = (__bf16*)(ws + (size_t)NT * n_nodes * sizeof(int)
                               + (size_t)NT * n_nodes * D * sizeof(__bf16));

    const int tot_cnt = NT * n_nodes;
    rgcn_zero_i32<<<(tot_cnt + 255) / 256, 256, 0, stream>>>(cnt, tot_cnt);

    const int n4 = n_nodes * D / 4;
    rgcn_cvt_bf16<<<(n4 + 255) / 256, 256, 0, stream>>>(x, xb, n4);

    const int tot_e = NT * n_edges;
    rgcn_count<<<(tot_e + 255) / 256, 256, 0, stream>>>(dst, cnt, tot_e,
                                                        n_nodes, n_edges);

    const int n_groups = n_nodes / TM;             // 6250
    dim3 ggrid(625, NT + 1);                       // y = etype 0..3 + self
    rgcn_gemm<<<ggrid, 256, 0, stream>>>(xb, W, b, W_self, b_self, wx, out,
                                         n_nodes, n_groups);

    // 8 wave32 edges per 256-thread block
    rgcn_scatter<<<(tot_e + 7) / 8, 256, 0, stream>>>(wx, src, dst, cnt, out,
                                                      tot_e, n_nodes, n_edges);
}
